// Encoder_22436909154571
// MI455X (gfx1250) — matrix operand order repaired and checked by reading, not compiled, not run on hardware
//
#include <hip/hip_runtime.h>
#include <hip/hip_bf16.h>

#define D_MODEL 128
#define NHEAD   4
#define DKH     32
#define BATCH   16
#define SEQ     2048
#define NTOK    (BATCH*SEQ)
#define EPSLN   1e-5f

typedef __attribute__((ext_vector_type(16))) _Float16 v16h;
typedef __attribute__((ext_vector_type(8)))  float    v8f;
typedef __attribute__((ext_vector_type(4)))  unsigned v4u;
typedef __attribute__((ext_vector_type(8)))  unsigned v8u;

// ---------------- cross-lane reductions via ds_swizzle (wave32) ------------
#define SWZ_XOR_F(v, pat) __int_as_float(__builtin_amdgcn_ds_swizzle(__float_as_int(v), (pat)))

__device__ __forceinline__ float red_max16(float v) {            // xor 1,2,4,8: closed in 16-lane halves
  v = fmaxf(v, SWZ_XOR_F(v, 0x041F));
  v = fmaxf(v, SWZ_XOR_F(v, 0x081F));
  v = fmaxf(v, SWZ_XOR_F(v, 0x101F));
  v = fmaxf(v, SWZ_XOR_F(v, 0x201F));
  return v;
}
__device__ __forceinline__ float red_sum16(float v) {
  v += SWZ_XOR_F(v, 0x041F);
  v += SWZ_XOR_F(v, 0x081F);
  v += SWZ_XOR_F(v, 0x101F);
  v += SWZ_XOR_F(v, 0x201F);
  return v;
}
__device__ __forceinline__ float red_pair32(float v) {           // combine lane L with L^16
  return v + SWZ_XOR_F(v, 0x401F);
}

// ---------------- WMMA fragment helpers (contiguous k = 16*(lane/16) + s in A and B) ----------------
// A 16x32 f16, row-major source with leading dim ld. lane = M (mod 16),
// kb = lane<16 ? 0 : 16;  VGPR v(0..3): K = kb+2v,+1 ; VGPR v(4..7): K = kb+8+2(v-4),+1
__device__ __forceinline__ v16h load_a_frag(const _Float16* __restrict__ base, int ld) {
  int lane = threadIdx.x & 31;
  int m  = lane & 15;
  int kb = (lane & 16) ? 16 : 0;
  const _Float16* p = base + m * ld;
  v16h a;
#pragma unroll
  for (int v = 0; v < 4; ++v) { int k = kb + 2*v;      a[2*v]   = p[k]; a[2*v+1]   = p[k+1]; }
#pragma unroll
  for (int v = 0; v < 4; ++v) { int k = kb + 8 + 2*v; a[8+2*v] = p[k]; a[8+2*v+1] = p[k+1]; }
  return a;
}
// B 32x16 f16 where source is stored [N][K] row-major (i.e. B[k][n] = src[n*ld + k]).
// lane = N, kb = lane<16 ? 0 : 16; VGPR v: K = kb+2v,+1  -> 16 contiguous halves per lane.
__device__ __forceinline__ v16h load_b_frag_nk(const _Float16* __restrict__ base, int ld) {
  int lane = threadIdx.x & 31;
  int n  = lane & 15;
  int kb = (lane & 16) ? 16 : 0;
  const _Float16* p = base + n * ld + kb;
  v16h b;
#pragma unroll
  for (int v = 0; v < 16; ++v) b[v] = p[v];
  return b;
}
// B 32x16 f16 where source is stored [K][N] row-major (B[k][n] = src[k*ld + n]).
__device__ __forceinline__ v16h load_b_frag_kn(const _Float16* __restrict__ base, int ld) {
  int lane = threadIdx.x & 31;
  int n  = lane & 15;
  int kb = (lane & 16) ? 16 : 0;
  v16h b;
#pragma unroll
  for (int v = 0; v < 8; ++v) {
    b[2*v]   = base[(kb + 2*v)     * ld + n];
    b[2*v+1] = base[(kb + 2*v + 1) * ld + n];
  }
  return b;
}

__device__ __forceinline__ v8f wmma_f16(v16h a, v16h b, v8f c) {
  return __builtin_amdgcn_wmma_f32_16x16x32_f16(false, a, false, b, (short)0, c, false, false);
}

// ======================= Kernel 1: LayerNorm(x) -> xn (f32) + xnh (f16) =====
__global__ void __launch_bounds__(256)
k_ln1(const float* __restrict__ x, const float* __restrict__ lnw,
      const float* __restrict__ lnb, float* __restrict__ xn,
      _Float16* __restrict__ xnh) {
  int tid  = threadIdx.x;
  int lane = tid & 31, w = tid >> 5;
  int r    = blockIdx.x * 128 + w * 16 + (lane & 15);   // token
  int hi   = (lane >> 4) & 1;                           // which 64-elem half
  const float4* px = (const float4*)(x + (size_t)r * D_MODEL + hi * 64);
  float4 buf[16];
  float s = 0.f, sq = 0.f;
#pragma unroll
  for (int i = 0; i < 16; ++i) {
    float4 v = px[i]; buf[i] = v;
    s  += v.x + v.y + v.z + v.w;
    sq += v.x*v.x + v.y*v.y + v.z*v.z + v.w*v.w;
  }
  s  = red_pair32(s);
  sq = red_pair32(sq);
  float mu   = s * (1.f / 128.f);
  float var  = sq * (1.f / 128.f) - mu * mu;
  float rinv = rsqrtf(var + EPSLN);
  float*    pxn = xn  + (size_t)r * D_MODEL + hi * 64;
  _Float16* pxh = xnh + (size_t)r * D_MODEL + hi * 64;
#pragma unroll
  for (int i = 0; i < 16; ++i) {
    float4 v = buf[i];
    int d0 = hi * 64 + i * 4;
    float o0 = (v.x - mu) * rinv * lnw[d0+0] + lnb[d0+0];
    float o1 = (v.y - mu) * rinv * lnw[d0+1] + lnb[d0+1];
    float o2 = (v.z - mu) * rinv * lnw[d0+2] + lnb[d0+2];
    float o3 = (v.w - mu) * rinv * lnw[d0+3] + lnb[d0+3];
    pxn[i*4+0] = o0; pxn[i*4+1] = o1; pxn[i*4+2] = o2; pxn[i*4+3] = o3;
    pxh[i*4+0] = (_Float16)o0; pxh[i*4+1] = (_Float16)o1;
    pxh[i*4+2] = (_Float16)o2; pxh[i*4+3] = (_Float16)o3;
  }
}

// ============ Kernel 2: q=relu(xn@wq^T), k=xn@wk^T, v=xn@wv^T (f16 out) =====
__global__ void __launch_bounds__(256)
k_qkv(const _Float16* __restrict__ xnh,
      const float* __restrict__ wq, const float* __restrict__ wk,
      const float* __restrict__ wv,
      _Float16* __restrict__ qh, _Float16* __restrict__ kh,
      _Float16* __restrict__ vh) {
  __shared__ __align__(16) _Float16 wlds[3 * D_MODEL * D_MODEL];   // 96 KB
  int tid = threadIdx.x;
  for (int i = tid; i < 3 * D_MODEL * D_MODEL; i += 256) {
    int which = i >> 14, idx = i & 16383;
    const float* src = (which == 0) ? wq : (which == 1) ? wk : wv;
    wlds[i] = (_Float16)src[idx];
  }
  __syncthreads();

  int lane = tid & 31, w = tid >> 5;
  int tbase = blockIdx.x * 128 + w * 16;
  v16h a4[4];
#pragma unroll
  for (int kt = 0; kt < 4; ++kt)
    a4[kt] = load_a_frag(xnh + (size_t)tbase * D_MODEL + kt * 32, D_MODEL);

  int n = lane & 15, hi8 = (lane & 16) ? 8 : 0;
  _Float16* outs[3] = {qh, kh, vh};
#pragma unroll
  for (int o = 0; o < 3; ++o) {
    for (int nt = 0; nt < 8; ++nt) {
      v8f acc = {};
#pragma unroll
      for (int kt = 0; kt < 4; ++kt) {
        v16h b = load_b_frag_nk(wlds + o * 16384 + (nt * 16) * D_MODEL + kt * 32, D_MODEL);
        acc = wmma_f16(a4[kt], b, acc);
      }
      int col = nt * 16 + n;
      int h = col >> 5, dk = col & 31;
#pragma unroll
      for (int v = 0; v < 8; ++v) {
        int tok  = tbase + v + hi8;
        int bb   = tok >> 11, srow = tok & (SEQ - 1);
        float val = acc[v];
        if (o == 0) val = fmaxf(val, 0.f);                       // relu-gated Q
        outs[o][((size_t)(bb * NHEAD + h) * SEQ + srow) * DKH + dk] = (_Float16)val;
      }
    }
  }
}

// ====== Kernel 3: flash attention per (b,h); 8 waves share K/V via LDS ======
// K tile staged by the Tensor Data Mover (tensor_load_to_lds, TENSORcnt),
// V tile staged by per-lane async copies (global_load_async_to_lds_b128, ASYNCcnt).
__global__ void __launch_bounds__(256)
k_attn(const _Float16* __restrict__ qh, const _Float16* __restrict__ kh,
       const _Float16* __restrict__ vh, const int* __restrict__ mask,
       const float* __restrict__ xn, float* __restrict__ sa) {
  __shared__ __align__(16) _Float16 kt_lds[32 * DKH];
  __shared__ __align__(16) _Float16 vt_lds[32 * DKH];
  __shared__ __align__(16) _Float16 p_lds[8 * 16 * 32];
  __shared__ float mask_lds[SEQ];

  int tid = threadIdx.x, lane = tid & 31, w = tid >> 5;
  const int per_bh = SEQ / 128;                    // 16 blocks per (b,h)
  int bh = blockIdx.x / per_bh;
  int qc = blockIdx.x % per_bh;
  int b  = bh / NHEAD;
  int q0 = qc * 128 + w * 16;

  for (int j = tid; j < SEQ; j += 256)
    mask_lds[j] = (mask[b * SEQ + j] == 0) ? -3e38f : 0.0f;      // additive key mask
  __syncthreads();

  const _Float16* kbase = kh + (size_t)bh * SEQ * DKH;
  const _Float16* vbase = vh + (size_t)bh * SEQ * DKH;
  v16h aq = load_a_frag(qh + (size_t)(bh * SEQ + q0) * DKH, DKH); // full K=DK=32

  // LDS byte addresses (low 32 bits of the flat address = allocation offset)
  unsigned ktl = (unsigned)(size_t)(void*)kt_lds;
  unsigned vtl = (unsigned)(size_t)(void*)vt_lds;

  v8f accO[2] = {};
  float m_i[8], l_i[8];
#pragma unroll
  for (int v = 0; v < 8; ++v) { m_i[v] = -3e38f; l_i[v] = 0.f; }
  int n = lane & 15, hi8 = (lane & 16) ? 8 : 0;
  const float scale = 0.17677669529663687f;        // 1/sqrt(32)

  for (int j0 = 0; j0 < SEQ; j0 += 32) {
    __syncthreads();                               // protect K/V tiles from prev iter readers
    if (w == 0) {
      // ---- K tile (2 KB contiguous) via Tensor Data Mover ----
      unsigned long long ga = (unsigned long long)(size_t)kbase +
                              (unsigned long long)j0 * (DKH * 2ull);
      // D# group 0: count=1 | lds_addr | global_addr[56:0] | type=2
      v4u g0 = { 1u, ktl, (unsigned)ga,
                 (unsigned)((ga >> 32) & 0x1FFFFFFu) | (2u << 30) };
      // D# group 1: data_size=3 (8B elems); tensor_dim0=256; tensor_dim1=1;
      // tile_dim0=256 (1-D tile, tile_dim1=0); tensor_dim0_stride=256
      v8u g1 = { 3u << 16,          // workgroup_mask=0, data_size=3
                 256u << 16,        // tensor_dim0[15:0] @ bits 63:48
                 1u << 16,          // tensor_dim0 hi=0, tensor_dim1=1
                 256u << 16,        // tile_dim0=256 @ bits 127:112
                 0u,                // tile_dim1=0, tile_dim2=0
                 256u,              // tensor_dim0_stride lo32
                 0u, 0u };
      asm volatile("tensor_load_to_lds %0, %1" :: "s"(g0), "s"(g1) : "memory");
      __builtin_amdgcn_s_wait_tensorcnt(0);
    } else if (w >= 4) {
      // ---- V tile (2 KB) via async direct-to-LDS loads, 16 B per lane ----
      int i = tid & 127;
      unsigned ldsa = vtl + (unsigned)i * 16u;
      unsigned goff = (unsigned)j0 * (DKH * 2u) + (unsigned)i * 16u;
      asm volatile("global_load_async_to_lds_b128 %0, %1, %2"
                   :: "v"(ldsa), "v"(goff),
                      "s"((unsigned long long)(size_t)vbase)
                   : "memory");
      asm volatile("s_wait_asynccnt 0x0" ::: "memory");
    }
    __syncthreads();

    float sc[2][8];
#pragma unroll
    for (int t = 0; t < 2; ++t) {
      v16h bk = load_b_frag_nk(kt_lds + (t * 16) * DKH, DKH);    // B[k=d][n=key]
      v8f z = {};
      v8f s8 = wmma_f16(aq, bk, z);
      float madd = mask_lds[j0 + t * 16 + n];
#pragma unroll
      for (int v = 0; v < 8; ++v) sc[t][v] = s8[v] * scale + madd;
    }

    _Float16* pw = p_lds + w * 512;                // this wave's 16x32 P tile
#pragma unroll
    for (int v = 0; v < 8; ++v) {                  // online softmax, row m = v + hi8
      float mx = red_max16(fmaxf(sc[0][v], sc[1][v]));
      float nm = fmaxf(m_i[v], mx);
      float p0 = __expf(sc[0][v] - nm);
      float p1 = __expf(sc[1][v] - nm);
      float rs = red_sum16(p0 + p1);
      float alpha = __expf(m_i[v] - nm);
      m_i[v] = nm;
      l_i[v] = l_i[v] * alpha + rs;
      accO[0][v] *= alpha;
      accO[1][v] *= alpha;
      pw[(v + hi8) * 32 + n]      = (_Float16)p0;
      pw[(v + hi8) * 32 + 16 + n] = (_Float16)p1;
    }
    v16h ap = load_a_frag(pw, 32);                 // C-layout -> A-layout via LDS bounce
#pragma unroll
    for (int t2 = 0; t2 < 2; ++t2) {
      v16h bv = load_b_frag_kn(vt_lds + t2 * 16, DKH);           // B[k=key][n=dk]
      accO[t2] = wmma_f16(ap, bv, accO[t2]);
    }
  }

  int h = bh % NHEAD;
#pragma unroll
  for (int t2 = 0; t2 < 2; ++t2) {
    int d = h * DKH + t2 * 16 + n;
#pragma unroll
    for (int v = 0; v < 8; ++v) {
      int srow = q0 + v + hi8;
      float o = accO[t2][v] / l_i[v];
      size_t idx = ((size_t)b * SEQ + srow) * D_MODEL + d;
      sa[idx] = xn[idx] + o;                       // residual: xn + attn
    }
  }
}

// ============ Kernel 4: h=LN(sa); out = sa + relu(h@w1^T+b1)@w2^T+b2 ========
__global__ void __launch_bounds__(256)
k_ffn(const float* __restrict__ sa, const float* __restrict__ lnw,
      const float* __restrict__ lnb, const float* __restrict__ w1,
      const float* __restrict__ b1, const float* __restrict__ w2,
      const float* __restrict__ b2, float* __restrict__ out) {
  __shared__ __align__(16) _Float16 w1l[D_MODEL * D_MODEL];
  __shared__ __align__(16) _Float16 w2l[D_MODEL * D_MODEL];
  __shared__ float bl[2 * D_MODEL];
  __shared__ __align__(16) _Float16 hl [8 * 16 * D_MODEL];
  __shared__ __align__(16) _Float16 h2l[8 * 16 * D_MODEL];

  int tid = threadIdx.x, lane = tid & 31, w = tid >> 5;
  for (int i = tid; i < D_MODEL * D_MODEL; i += 256) {
    w1l[i] = (_Float16)w1[i];
    w2l[i] = (_Float16)w2[i];
  }
  if (tid < 128) bl[tid] = b1[tid]; else bl[tid] = b2[tid - 128];

  int tbase = blockIdx.x * 128 + w * 16;
  int r = lane & 15, hi = (lane >> 4) & 1;
  int tok = tbase + r;
  const float4* ps = (const float4*)(sa + (size_t)tok * D_MODEL + hi * 64);
  float4 buf[16];
  float s = 0.f, sq = 0.f;
#pragma unroll
  for (int i = 0; i < 16; ++i) {
    float4 v = ps[i]; buf[i] = v;
    s  += v.x + v.y + v.z + v.w;
    sq += v.x*v.x + v.y*v.y + v.z*v.z + v.w*v.w;
  }
  s  = red_pair32(s);
  sq = red_pair32(sq);
  float mu   = s * (1.f / 128.f);
  float var  = sq * (1.f / 128.f) - mu * mu;
  float rinv = rsqrtf(var + EPSLN);
  _Float16* hrow = hl + w * 2048 + r * D_MODEL + hi * 64;
#pragma unroll
  for (int i = 0; i < 16; ++i) {
    float4 v = buf[i];
    int d0 = hi * 64 + i * 4;
    hrow[i*4+0] = (_Float16)((v.x - mu) * rinv * lnw[d0+0] + lnb[d0+0]);
    hrow[i*4+1] = (_Float16)((v.y - mu) * rinv * lnw[d0+1] + lnb[d0+1]);
    hrow[i*4+2] = (_Float16)((v.z - mu) * rinv * lnw[d0+2] + lnb[d0+2]);
    hrow[i*4+3] = (_Float16)((v.w - mu) * rinv * lnw[d0+3] + lnb[d0+3]);
  }
  __syncthreads();

  int n = lane & 15, hi8 = (lane & 16) ? 8 : 0;
  v16h a1[4];
#pragma unroll
  for (int kt = 0; kt < 4; ++kt)
    a1[kt] = load_a_frag(hl + w * 2048 + kt * 32, D_MODEL);
  for (int nt = 0; nt < 8; ++nt) {
    v8f acc = {};
#pragma unroll
    for (int kt = 0; kt < 4; ++kt)
      acc = wmma_f16(a1[kt], load_b_frag_nk(w1l + nt * 16 * D_MODEL + kt * 32, D_MODEL), acc);
    int col = nt * 16 + n;
    float bias = bl[col];
#pragma unroll
    for (int v = 0; v < 8; ++v)
      h2l[w * 2048 + (v + hi8) * D_MODEL + col] = (_Float16)fmaxf(acc[v] + bias, 0.f);
  }
  __syncthreads();

  v16h a2[4];
#pragma unroll
  for (int kt = 0; kt < 4; ++kt)
    a2[kt] = load_a_frag(h2l + w * 2048 + kt * 32, D_MODEL);
  for (int nt = 0; nt < 8; ++nt) {
    v8f acc = {};
#pragma unroll
    for (int kt = 0; kt < 4; ++kt)
      acc = wmma_f16(a2[kt], load_b_frag_nk(w2l + nt * 16 * D_MODEL + kt * 32, D_MODEL), acc);
    int col = nt * 16 + n;
    float bias = bl[128 + col];
#pragma unroll
    for (int v = 0; v < 8; ++v) {
      int tk = tbase + v + hi8;
      size_t idx = (size_t)tk * D_MODEL + col;
      out[idx] = sa[idx] + acc[v] + bias;          // final residual
    }
  }
}

// =========================== host-side launcher =============================
extern "C" void kernel_launch(void* const* d_in, const int* in_sizes, int n_in,
                              void* d_out, int out_size, void* d_ws, size_t ws_size,
                              hipStream_t stream) {
  const float* x    = (const float*)d_in[0];
  const int*   mask = (const int*)  d_in[1];
  const float* lnw  = (const float*)d_in[2];
  const float* lnb  = (const float*)d_in[3];
  const float* wq   = (const float*)d_in[4];
  const float* wk   = (const float*)d_in[5];
  const float* wv   = (const float*)d_in[6];
  const float* w1   = (const float*)d_in[7];
  const float* b1   = (const float*)d_in[8];
  const float* w2   = (const float*)d_in[9];
  const float* b2   = (const float*)d_in[10];
  float* out = (float*)d_out;

  // workspace layout (64 MB total)
  float*    xn  = (float*)d_ws;                               // 16 MB f32
  _Float16* xnh = (_Float16*)(xn + (size_t)NTOK * D_MODEL);   //  8 MB f16
  _Float16* qh  = xnh + (size_t)NTOK * D_MODEL;               //  8 MB f16 [B,H,S,DK]
  _Float16* kh  = qh  + (size_t)NTOK * D_MODEL;
  _Float16* vh  = kh  + (size_t)NTOK * D_MODEL;
  float*    sa  = (float*)(vh + (size_t)NTOK * D_MODEL);      // 16 MB f32

  k_ln1 <<<NTOK / 128, 256, 0, stream>>>(x, lnw, lnb, xn, xnh);
  k_qkv <<<NTOK / 128, 256, 0, stream>>>(xnh, wq, wk, wv, qh, kh, vh);
  k_attn<<<BATCH * NHEAD * (SEQ / 128), 256, 0, stream>>>(qh, kh, vh, mask, xn, sa);
  k_ffn <<<NTOK / 128, 256, 0, stream>>>(sa, lnw, lnb, w1, b1, w2, b2, out);
}
